// CPRRouter_28003186770655
// MI455X (gfx1250) — matrix operand order repaired and checked by reading, not compiled, not run on hardware
//
#include <hip/hip_runtime.h>
#include <hip/hip_fp16.h>
#include <math.h>

typedef __attribute__((ext_vector_type(16))) _Float16 v16h;
typedef __attribute__((ext_vector_type(8)))  float    v8f;

#define T_TOKENS 16384
#define DIM      2048
#define NEXP     64
#define TOPK     8

// ---------------------------------------------------------------------------
// Kernel 1: L2-normalize the 64 expert prototypes, emit f16 [64][2048] into ws
// ---------------------------------------------------------------------------
__global__ __launch_bounds__(256)
void proto_norm_kernel(const float* __restrict__ proto, _Float16* __restrict__ pf16) {
  __shared__ float red[256];
  const int e   = blockIdx.x;
  const int tid = threadIdx.x;
  const float* row = proto + (size_t)e * DIM;

  float ss = 0.f;
  for (int i = tid; i < DIM; i += 256) { float v = row[i]; ss += v * v; }
  red[tid] = ss;
  __syncthreads();
  for (int s = 128; s > 0; s >>= 1) {
    if (tid < s) red[tid] += red[tid + s];
    __syncthreads();
  }
  const float scale = 1.0f / fmaxf(sqrtf(red[0]), 1e-12f);
  for (int i = tid; i < DIM; i += 256)
    pf16[(size_t)e * DIM + i] = (_Float16)(row[i] * scale);
}

// ---------------------------------------------------------------------------
// Kernel 2: logits via v_wmma_f32_16x16x32_f16, softmax(64), top-8
//   block = 256 threads = 8 wave32, 128 tokens/block, grid = 128 blocks
//   wave tile: 16 tokens x 64 experts (4 WMMA accumulators)
//   proto K-slices staged with GLOBAL_LOAD_ASYNC_TO_LDS_B128 (ASYNCcnt path)
// ---------------------------------------------------------------------------
__global__ __launch_bounds__(256)
void router_kernel(const float* __restrict__ hs,
                   const _Float16* __restrict__ pf16,
                   float* __restrict__ outw, int* __restrict__ outi) {
  // proto K-slice, padded rows: 272B stride -> conflict-free ds_load_b128
  __shared__ _Float16 pbuf[NEXP][136];
  __shared__ float probs[8][16][64];

  const int tid  = threadIdx.x;
  const int lane = tid & 31;
  const int wv   = tid >> 5;
  const int m    = lane & 15;   // token row (A) / expert col (B) within tile
  const int hi   = lane >> 4;   // half-wave selector in WMMA lane layout

  const int tokBase = blockIdx.x * 128 + wv * 16;
  const float* arow = hs + (size_t)(tokBase + m) * DIM;

  v8f acc[4] = {v8f{}, v8f{}, v8f{}, v8f{}};
  float ss = 0.f;

  for (int k0 = 0; k0 < DIM; k0 += 128) {
    // prefetch next A slice while we work on this one
    if (k0 + 128 < DIM) __builtin_prefetch(arow + k0 + 128, 0, 1);

    __syncthreads();   // protect pbuf from readers of previous slice

    // --- stage proto[:, k0:k0+128) into LDS via async DMA (no VGPR bounce) --
    // Each lane copies 16B: 256 threads x 4 iters = 16 KB slice.
    #pragma unroll
    for (int it = 0; it < 4; ++it) {
      const int li = it * 256 + tid;      // 1024 groups of 8 halves
      const int ex = li >> 4;             // 16 groups of 8 per expert row
      const int kg = (li & 15) * 8;
      const _Float16* g = pf16 + (size_t)ex * DIM + k0 + kg;
      const unsigned ldsoff = (unsigned)(size_t)(&pbuf[ex][kg]); // wave-rel LDS byte offset
      asm volatile("global_load_async_to_lds_b128 %0, %1, off"
                   :: "v"(ldsoff), "v"(g) : "memory");
    }
    asm volatile("s_wait_asynccnt 0x0" ::: "memory");
    __syncthreads();

    #pragma unroll
    for (int c4 = 0; c4 < 4; ++c4) {
      const int kb = k0 + c4 * 32;
      // ---- A fragment: 16-bit A 16x32 layout ----
      // lane(l<16):  e0..7 -> K=kb+0..7,   e8..15 -> K=kb+16..23
      // lane(l>=16): e0..7 -> K=kb+8..15,  e8..15 -> K=kb+24..31
      const float4 f0 = *(const float4*)(arow + kb + hi * 8);
      const float4 f1 = *(const float4*)(arow + kb + hi * 8 + 4);
      const float4 f2 = *(const float4*)(arow + kb + 16 + hi * 8);
      const float4 f3 = *(const float4*)(arow + kb + 16 + hi * 8 + 4);

      ss += f0.x*f0.x + f0.y*f0.y + f0.z*f0.z + f0.w*f0.w
          + f1.x*f1.x + f1.y*f1.y + f1.z*f1.z + f1.w*f1.w
          + f2.x*f2.x + f2.y*f2.y + f2.z*f2.z + f2.w*f2.w
          + f3.x*f3.x + f3.y*f3.y + f3.z*f3.z + f3.w*f3.w;

      v16h a;
      a[0]=(_Float16)f0.x; a[1]=(_Float16)f0.y; a[2]=(_Float16)f0.z; a[3]=(_Float16)f0.w;
      a[4]=(_Float16)f1.x; a[5]=(_Float16)f1.y; a[6]=(_Float16)f1.z; a[7]=(_Float16)f1.w;
      a[8]=(_Float16)f2.x; a[9]=(_Float16)f2.y; a[10]=(_Float16)f2.z; a[11]=(_Float16)f2.w;
      a[12]=(_Float16)f3.x; a[13]=(_Float16)f3.y; a[14]=(_Float16)f3.z; a[15]=(_Float16)f3.w;

      // ---- batch all 4 B fragments (8x ds_load_b128), then WMMA quartet ----
      union Bf { v16h v; float4 f[2]; } b[4];
      #pragma unroll
      for (int n = 0; n < 4; ++n) {
        const int ex = n * 16 + m;
        b[n].f[0] = *(const float4*)(&pbuf[ex][c4 * 32 + hi * 8]);
        b[n].f[1] = *(const float4*)(&pbuf[ex][c4 * 32 + 16 + hi * 8]);
      }
      #pragma unroll
      for (int n = 0; n < 4; ++n) {
        acc[n] = __builtin_amdgcn_wmma_f32_16x16x32_f16(
            false, a, false, b[n].v, (short)0, acc[n], false, false);
      }
    }
  }

  // ss partial coverage: lane l and lane l^16 together cover row m fully
  ss += __shfl_xor(ss, 16, 32);   // now every lane holds ss of row (lane&15)

  // ---- scale by 1/||h||, softmax over 64 experts, write probs to LDS ----
  #pragma unroll
  for (int r = 0; r < 8; ++r) {
    const int t = r + 8 * hi;                       // wave-local token of this D row
    const float ssr   = __shfl(ss, t, 32);          // lane t holds ss of row t
    const float scale = 1.0f / fmaxf(sqrtf(ssr), 1e-12f);

    float v0 = acc[0][r] * scale;
    float v1 = acc[1][r] * scale;
    float v2 = acc[2][r] * scale;
    float v3 = acc[3][r] * scale;

    float mx = fmaxf(fmaxf(v0, v1), fmaxf(v2, v3));
    #pragma unroll
    for (int off = 1; off < 16; off <<= 1)
      mx = fmaxf(mx, __shfl_xor(mx, off, 32));      // stays within half-wave

    float e0 = __expf(v0 - mx), e1 = __expf(v1 - mx);
    float e2 = __expf(v2 - mx), e3 = __expf(v3 - mx);
    float sm = e0 + e1 + e2 + e3;
    #pragma unroll
    for (int off = 1; off < 16; off <<= 1)
      sm += __shfl_xor(sm, off, 32);
    const float inv = 1.0f / sm;

    probs[wv][t][ 0 + m] = e0 * inv;
    probs[wv][t][16 + m] = e1 * inv;
    probs[wv][t][32 + m] = e2 * inv;
    probs[wv][t][48 + m] = e3 * inv;
  }
  __syncthreads();

  // ---- top-8 of 64 per token (first-max scan => lowest-index tie-break) ----
  if (lane < 16) {
    const int gt = tokBase + lane;
    float* p = probs[wv][lane];
    for (int j = 0; j < TOPK; ++j) {
      float best = -1.0f; int bi = 0;
      #pragma unroll 8
      for (int e = 0; e < NEXP; ++e) {
        const float v = p[e];
        if (v > best) { best = v; bi = e; }
      }
      outw[(size_t)gt * TOPK + j] = best;
      outi[(size_t)gt * TOPK + j] = bi;
      p[bi] = -2.0f;
    }
  }
}

// ---------------------------------------------------------------------------
extern "C" void kernel_launch(void* const* d_in, const int* in_sizes, int n_in,
                              void* d_out, int out_size, void* d_ws, size_t ws_size,
                              hipStream_t stream) {
  const float* hs    = (const float*)d_in[0];   // [16384, 2048] f32
  const float* proto = (const float*)d_in[1];   // [64, 2048] f32
  _Float16* pf16 = (_Float16*)d_ws;             // 64*2048*2 = 256 KB scratch

  float* outw = (float*)d_out;                            // [16384,8] f32
  int*   outi = (int*)(outw + (size_t)T_TOKENS * TOPK);   // [16384,8] i32

  proto_norm_kernel<<<NEXP, 256, 0, stream>>>(proto, pf16);
  router_kernel<<<T_TOKENS / 128, 256, 0, stream>>>(hs, pf16, outw, outi);
}
